// TwistorLNNBlock_45183055954285
// MI455X (gfx1250) — compile-verified, matrix operand order checked by reading, not run on hardware
//
#include <hip/hip_runtime.h>
#include <hip/hip_bf16.h>
#include <math.h>

#define HH   1024
#define DIN  512
#define TT   512
#define BB   32
#define DT_F 0.1f
#define NBLK 128   // persistent grid: 128 tiles (2 Mtiles x 64 Ntiles), 1 tile/block
#define WROW 1032  // padded LDS weight row stride in bf16 (2064 B = 516 dwords -> 4-bank skew)

typedef __attribute__((ext_vector_type(16))) __bf16 bf16x16;
typedef __attribute__((ext_vector_type(8)))  __bf16 bf16x8;
typedef __attribute__((ext_vector_type(8)))  float  f32x8;

#define WMMA_BF16(a, b, c) \
  __builtin_amdgcn_wmma_f32_16x16x32_bf16(false, (a), false, (b), (short)0, (c), false, false)

// Load one 16x32 bf16 fragment from global (A layout; also serves B of W^T
// since B[k,n] = W[n,k] with W row-major).
// CDNA5 16-bit A-matrix table: lanes 0-15 hold row (lane&15),
// K = {k0..k0+7, k0+16..k0+23}; lanes 16-31 hold K = {k0+8..k0+15, k0+24..k0+31}.
__device__ __forceinline__ bf16x16 load_frag(const __bf16* base, int row0, int ld,
                                             int k0, int lane) {
  int r    = row0 + (lane & 15);
  int koff = k0 + ((lane & 16) ? 8 : 0);
  const __bf16* p = base + (size_t)r * ld + koff;
  bf16x8 lo = *(const bf16x8*)(p);
  bf16x8 hi = *(const bf16x8*)(p + 16);
  return __builtin_shufflevector(lo, hi, 0,1,2,3,4,5,6,7,8,9,10,11,12,13,14,15);
}

__device__ __forceinline__ float sigmoidf_(float x) {
  return 1.0f / (1.0f + __expf(-x));
}

// ---- prep kernels -----------------------------------------------------------

__global__ void prep_weights_k(const float* __restrict__ Wr_w, const float* __restrict__ mask_r,
                               const float* __restrict__ Wi_w, const float* __restrict__ mask_i,
                               const float* __restrict__ Wt_w, const float* __restrict__ out_w,
                               __bf16* wr, __bf16* wi, __bf16* wt, __bf16* ow) {
  int i = blockIdx.x * blockDim.x + threadIdx.x;
  if (i >= HH * HH) return;
  wr[i] = (__bf16)(Wr_w[i] * sigmoidf_(mask_r[i]));
  wi[i] = (__bf16)(Wi_w[i] * sigmoidf_(mask_i[i]));
  wt[i] = (__bf16)(Wt_w[i]);
  ow[i] = (__bf16)(out_w[i]);
}

__global__ void prep_u_k(const float* __restrict__ U_w, __bf16* ub) {
  int i = blockIdx.x * blockDim.x + threadIdx.x;
  if (i < HH * DIN) ub[i] = (__bf16)U_w[i];
}

__global__ void prep_x_k(const float* __restrict__ x, __bf16* xb) {
  int i = blockIdx.x * blockDim.x + threadIdx.x;
  if (i < TT * BB * DIN) xb[i] = (__bf16)x[i];
}

__global__ void init_state_k(float* zr, float* zi,
                             __bf16* zb0, __bf16* zb1,
                             __bf16* ar0, __bf16* ai0, __bf16* am0,
                             __bf16* ar1, __bf16* ai1, __bf16* am1,
                             unsigned* bar) {
  int i = blockIdx.x * blockDim.x + threadIdx.x;
  if (i == 0) *bar = 0u;               // reset grid-barrier counter every call
  if (i >= BB * HH) return;
  zr[i] = 0.0f; zi[i] = 0.0f;
  __bf16 z = (__bf16)0.0f;
  zb0[i] = z; zb1[i] = z;
  ar0[i] = z; ai0[i] = z; am0[i] = z;
  ar1[i] = z; ai1[i] = z; am1[i] = z;
}

// ---- batched input projection: Ux = x @ U_w^T + U_b ------------------------
// M = T*B = 16384, N = H = 1024, K = D_IN = 512. One wave per 16x16 tile.
__global__ void ux_gemm_k(const __bf16* __restrict__ xb, const __bf16* __restrict__ ub,
                          const float* __restrict__ U_b, float* __restrict__ ux) {
  int wid  = (blockIdx.x * blockDim.x + threadIdx.x) >> 5;
  int lane = threadIdx.x & 31;
  int mt = wid >> 6;        // 1024 M-tiles
  int nt = wid & 63;        // 64 N-tiles
  int m0 = mt << 4, n0 = nt << 4;
  f32x8 c = {};
#pragma unroll 4
  for (int k0 = 0; k0 < DIN; k0 += 32) {
    bf16x16 a = load_frag(xb, m0, DIN, k0, lane);
    bf16x16 b = load_frag(ub, n0, DIN, k0, lane);
    c = WMMA_BF16(a, b, c);
  }
  int n = n0 + (lane & 15);
  float bias = U_b[n];
  int mbase = m0 + ((lane & 16) ? 8 : 0);
#pragma unroll
  for (int i = 0; i < 8; ++i)
    ux[(size_t)(mbase + i) * HH + n] = c[i] + bias;
}

// ---- persistent recurrence kernel -------------------------------------------
// The full T=512 scan lives in ONE kernel. 128 blocks (4 waves each); block b
// owns one 16x16 (batch x hidden) output tile across all steps. The block's
// weight slice (16 rows x 1024 K of Wr/Wi/Wt/out_w, 128 KB) is staged into LDS
// ONCE and served from ds_load_b128 for all 512 steps (CDNA5: 320 KB LDS/WGP).
// Per step, each wave accumulates a K=256 slice of four GEMMs:
//   cr = tanh(zr)@Wr^T, ci = tanh(zi)@Wi^T, ct = |z|@Wt^T, cy = zr_prev@out_w^T
// Partials reduce through LDS; wave 0 runs the coupled z-update epilogue and
// wave 1 emits the fused previous-step output y_{t-1}. Steps are separated by
// a device-scope grid barrier (monotonic-target atomic counter).
__global__ void recurrence_k(const __bf16* __restrict__ wr, const __bf16* __restrict__ wi,
                             const __bf16* __restrict__ wt, const __bf16* __restrict__ ow,
                             const float* __restrict__ Wr_b, const float* __restrict__ Wi_b,
                             const float* __restrict__ Wt_b,
                             const float* __restrict__ b_real, const float* __restrict__ b_imag,
                             const float* __restrict__ tau_bias, const float* __restrict__ out_b,
                             const float* __restrict__ ux_all,
                             float* __restrict__ zr, float* __restrict__ zi,
                             __bf16* ar0, __bf16* ai0, __bf16* am0, __bf16* zb0,
                             __bf16* ar1, __bf16* ai1, __bf16* am1, __bf16* zb1,
                             float* __restrict__ y_base, unsigned* bar) {
  __shared__ __bf16 wlds[4][16][WROW];  // 132 KB: LDS-resident weight slices
  __shared__ float  red[4][4][8][32];   // 16 KB: [split][acc][c-row][lane]

  const int wave = threadIdx.x >> 5;   // split-K index 0..3
  const int lane = threadIdx.x & 31;
  const int tile = blockIdx.x;         // 0..127
  const int mt = tile >> 6;            // 2 M-tiles (B = 32)
  const int nt = tile & 63;            // 64 N-tiles (H = 1024)
  const int m0 = mt << 4, n0 = nt << 4;
  const int kbeg = wave << 8;          // 256-wide K slice per wave

  const int n = n0 + (lane & 15);
  const int mbase = m0 + ((lane & 16) ? 8 : 0);

  // ---- one-time weight staging: global -> LDS (16 rows x 1024 halfs x 4) ----
  {
    const __bf16* wsrc[4] = { wr, wi, wt, ow };
#pragma unroll
    for (int m = 0; m < 4; ++m) {
      const __bf16* src = wsrc[m] + (size_t)n0 * HH;
      for (int u = threadIdx.x; u < 16 * 128; u += 128) {  // 16B units
        int row = u >> 7;
        int col = (u & 127) << 3;                          // halfs
        *(bf16x8*)&wlds[m][row][col] = *(const bf16x8*)(src + (size_t)row * HH + col);
      }
    }
  }
  __syncthreads();

  // B-fragment served from LDS (same lane pattern as load_frag).
  const int fr    = lane & 15;
  const int fkoff = (lane & 16) ? 8 : 0;
  auto ldsfrag = [&](int m, int k0) -> bf16x16 {
    int koff = k0 + fkoff;
    bf16x8 lo = *(const bf16x8*)&wlds[m][fr][koff];
    bf16x8 hi = *(const bf16x8*)&wlds[m][fr][koff + 16];
    return __builtin_shufflevector(lo, hi, 0,1,2,3,4,5,6,7,8,9,10,11,12,13,14,15);
  };

  // Loop-invariant per-column parameters (hoisted across all 512 steps).
  const float wrb = Wr_b[n], wib = Wi_b[n], wtb = Wt_b[n];
  const float brl = b_real[n], bim = b_imag[n], tbn = tau_bias[n];
  const float ob  = out_b[n];

  const unsigned nblk = gridDim.x;

  for (int t = 0; t < TT; ++t) {
    const int p = t & 1;
    const __bf16* ar_in = p ? ar1 : ar0;
    const __bf16* ai_in = p ? ai1 : ai0;
    const __bf16* am_in = p ? am1 : am0;
    const __bf16* zb_in = p ? zb1 : zb0;
    __bf16* ar_out = p ? ar0 : ar1;
    __bf16* ai_out = p ? ai0 : ai1;
    __bf16* am_out = p ? am0 : am1;
    __bf16* zb_out = p ? zb0 : zb1;
    const float* ux_t = ux_all + (size_t)t * BB * HH;

    f32x8 cr = {}, ci = {}, ct = {}, cy = {};
#pragma unroll 2
    for (int kk = 0; kk < 256; kk += 32) {
      int k0 = kbeg + kk;
      bf16x16 a0 = load_frag(ar_in, m0, HH, k0, lane);
      bf16x16 b0 = ldsfrag(0, k0);
      cr = WMMA_BF16(a0, b0, cr);
      bf16x16 a1 = load_frag(ai_in, m0, HH, k0, lane);
      bf16x16 b1 = ldsfrag(1, k0);
      ci = WMMA_BF16(a1, b1, ci);
      bf16x16 a2 = load_frag(am_in, m0, HH, k0, lane);
      bf16x16 b2 = ldsfrag(2, k0);
      ct = WMMA_BF16(a2, b2, ct);
      bf16x16 a3 = load_frag(zb_in, m0, HH, k0, lane);
      bf16x16 b3 = ldsfrag(3, k0);
      cy = WMMA_BF16(a3, b3, cy);
    }

#pragma unroll
    for (int i = 0; i < 8; ++i) {
      red[wave][0][i][lane] = cr[i];
      red[wave][1][i][lane] = ci[i];
      red[wave][2][i][lane] = ct[i];
      red[wave][3][i][lane] = cy[i];
    }
    __syncthreads();

    if (wave == 0) {
      // Reduce cr/ci/ct across splits, then coupled nonlinear state update.
#pragma unroll
      for (int i = 0; i < 8; ++i) {
        float gr = red[0][0][i][lane] + red[1][0][i][lane] + red[2][0][i][lane] + red[3][0][i][lane];
        float gi = red[0][1][i][lane] + red[1][1][i][lane] + red[2][1][i][lane] + red[3][1][i][lane];
        float gt = red[0][2][i][lane] + red[1][2][i][lane] + red[2][2][i][lane] + red[3][2][i][lane];
        int m = mbase + i;
        size_t idx = (size_t)m * HH + n;
        float zro = zr[idx], zio = zi[idx], u = ux_t[idx];
        float dr = -zro + gr + wrb + u + brl;
        float di = -zio + gi + wib + u + bim;
        float tau = sigmoidf_(gt + wtb) + tbn;
        tau = fminf(fmaxf(tau, 0.01f), 1.0f) + 1e-6f;
        dr = fminf(fmaxf(dr / tau, -10.0f), 10.0f);
        di = fminf(fmaxf(di / tau, -10.0f), 10.0f);
        float zrn = fminf(fmaxf(zro + DT_F * dr, -100.0f), 100.0f);
        float zin = fminf(fmaxf(zio + DT_F * di, -100.0f), 100.0f);
        zr[idx] = zrn;
        zi[idx] = zin;
        ar_out[idx] = (__bf16)tanhf(zrn);
        ai_out[idx] = (__bf16)tanhf(zin);
        am_out[idx] = (__bf16)sqrtf(zrn * zrn + zin * zin);
        zb_out[idx] = (__bf16)zrn;
      }
    } else if (wave == 1 && t > 0) {
      // Reduce cy and emit the fused previous-step output y_{t-1}.
      float* y = y_base + (size_t)(t - 1) * BB * HH;
#pragma unroll
      for (int i = 0; i < 8; ++i) {
        float gy = red[0][3][i][lane] + red[1][3][i][lane] + red[2][3][i][lane] + red[3][3][i][lane];
        y[(size_t)(mbase + i) * HH + n] = gy + ob;
      }
    }

    // ---- grid-wide barrier between steps ----
    // __syncthreads: all waves' stores issued/complete (storecnt) + LDS fence.
    __syncthreads();
    if (threadIdx.x == 0) {
      __threadfence();                                   // release: WGP$ writeback (DEV)
      __hip_atomic_fetch_add(bar, 1u, __ATOMIC_RELEASE, __HIP_MEMORY_SCOPE_AGENT);
      const unsigned target = nblk * (unsigned)(t + 1);  // monotonic, no reset race
      while (__hip_atomic_load(bar, __ATOMIC_ACQUIRE, __HIP_MEMORY_SCOPE_AGENT) < target) {
        __builtin_amdgcn_s_sleep(2);
      }
      __threadfence();                                   // acquire: WGP$ invalidate
    }
    __syncthreads();
  }
}

// ---- tail output for the final step: y_{T-1} = zr @ out_w^T + out_b --------
__global__ void y_gemm_k(const __bf16* __restrict__ zbf, const __bf16* __restrict__ ow,
                         const float* __restrict__ out_b, float* __restrict__ y) {
  int wid  = (blockIdx.x * blockDim.x + threadIdx.x) >> 5;
  int lane = threadIdx.x & 31;
  int mt = wid >> 6;
  int nt = wid & 63;
  int m0 = mt << 4, n0 = nt << 4;
  f32x8 c = {};
#pragma unroll 4
  for (int k0 = 0; k0 < HH; k0 += 32) {
    bf16x16 a = load_frag(zbf, m0, HH, k0, lane);
    bf16x16 b = load_frag(ow,  n0, HH, k0, lane);
    c = WMMA_BF16(a, b, c);
  }
  int n = n0 + (lane & 15);
  float bias = out_b[n];
  int mbase = m0 + ((lane & 16) ? 8 : 0);
#pragma unroll
  for (int i = 0; i < 8; ++i)
    y[(size_t)(mbase + i) * HH + n] = c[i] + bias;
}

// ---- launch -----------------------------------------------------------------

extern "C" void kernel_launch(void* const* d_in, const int* in_sizes, int n_in,
                              void* d_out, int out_size, void* d_ws, size_t ws_size,
                              hipStream_t stream) {
  const float* x_seq    = (const float*)d_in[0];
  const float* Wr_w     = (const float*)d_in[1];
  const float* Wr_b     = (const float*)d_in[2];
  const float* Wi_w     = (const float*)d_in[3];
  const float* Wi_b     = (const float*)d_in[4];
  const float* U_w      = (const float*)d_in[5];
  const float* U_b      = (const float*)d_in[6];
  const float* Wt_w     = (const float*)d_in[7];
  const float* Wt_b     = (const float*)d_in[8];
  const float* b_real   = (const float*)d_in[9];
  const float* b_imag   = (const float*)d_in[10];
  const float* mask_r   = (const float*)d_in[11];
  const float* mask_i   = (const float*)d_in[12];
  const float* tau_bias = (const float*)d_in[13];
  const float* out_w    = (const float*)d_in[14];
  const float* out_b    = (const float*)d_in[15];
  float* out = (float*)d_out;

  // Workspace carve-up (256-byte aligned slices).
  char* ws = (char*)d_ws;
  size_t off = 0;
  auto carve = [&](size_t bytes) -> char* {
    char* p = ws + off;
    off += (bytes + 255) & ~(size_t)255;
    return p;
  };
  __bf16* wr_bf  = (__bf16*)carve((size_t)HH * HH * 2);
  __bf16* wi_bf  = (__bf16*)carve((size_t)HH * HH * 2);
  __bf16* wt_bf  = (__bf16*)carve((size_t)HH * HH * 2);
  __bf16* ow_bf  = (__bf16*)carve((size_t)HH * HH * 2);
  __bf16* u_bf   = (__bf16*)carve((size_t)HH * DIN * 2);
  __bf16* x_bf   = (__bf16*)carve((size_t)TT * BB * DIN * 2);
  float*  ux     = (float*) carve((size_t)TT * BB * HH * 4);
  float*  zr     = (float*) carve((size_t)BB * HH * 4);
  float*  zi     = (float*) carve((size_t)BB * HH * 4);
  __bf16* zbf[2] = { (__bf16*)carve((size_t)BB * HH * 2), (__bf16*)carve((size_t)BB * HH * 2) };
  __bf16* ar[2]  = { (__bf16*)carve((size_t)BB * HH * 2), (__bf16*)carve((size_t)BB * HH * 2) };
  __bf16* ai[2]  = { (__bf16*)carve((size_t)BB * HH * 2), (__bf16*)carve((size_t)BB * HH * 2) };
  __bf16* am[2]  = { (__bf16*)carve((size_t)BB * HH * 2), (__bf16*)carve((size_t)BB * HH * 2) };
  unsigned* bar  = (unsigned*)carve(256);
  (void)ws_size; (void)n_in; (void)in_sizes; (void)out_size;

  // One-time prep.
  prep_weights_k<<<(HH * HH + 255) / 256, 256, 0, stream>>>(
      Wr_w, mask_r, Wi_w, mask_i, Wt_w, out_w, wr_bf, wi_bf, wt_bf, ow_bf);
  prep_u_k<<<(HH * DIN + 255) / 256, 256, 0, stream>>>(U_w, u_bf);
  prep_x_k<<<(TT * BB * DIN + 255) / 256, 256, 0, stream>>>(x_seq, x_bf);
  init_state_k<<<(BB * HH + 255) / 256, 256, 0, stream>>>(
      zr, zi, zbf[0], zbf[1], ar[0], ai[0], am[0], ar[1], ai[1], am[1], bar);

  // Batched input projection: (16384 x 512) @ (512 x 1024) -> 65536 tiles.
  ux_gemm_k<<<(1024 * 64) / 8, 256, 0, stream>>>(x_bf, u_bf, U_b, ux);

  // Entire sequential recurrence in ONE persistent kernel.
  recurrence_k<<<NBLK, 128, 0, stream>>>(
      wr_bf, wi_bf, wt_bf, ow_bf,
      Wr_b, Wi_b, Wt_b, b_real, b_imag, tau_bias, out_b,
      ux, zr, zi,
      ar[0], ai[0], am[0], zbf[0],
      ar[1], ai[1], am[1], zbf[1],
      out, bar);

  // Tail: y for the final step (state written to zbf[TT % 2] == zbf[0]).
  y_gemm_k<<<16, 256, 0, stream>>>(zbf[TT & 1], ow_bf, out_b,
                                   out + (size_t)(TT - 1) * BB * HH);
}
